// MHA_8040178778652
// MI455X (gfx1250) — matrix operand order repaired and checked by reading, not compiled, not run on hardware
//
#include <hip/hip_runtime.h>
#include <hip/hip_bf16.h>

typedef __attribute__((ext_vector_type(16))) _Float16 v16h;
typedef __attribute__((ext_vector_type(8)))  _Float16 v8h;
typedef __attribute__((ext_vector_type(8)))  float    v8f;
typedef __attribute__((ext_vector_type(4)))  int      v4i;

#define DIM      1024
#define HEADS    16
#define HEAD_DIM 64
#define SEQ      2048
#define BATCH    4
#define MROWS    (BATCH * SEQ)   // 8192 token rows

// ---------------------------------------------------------------------------
// CDNA5 async global->LDS copy (ASYNCcnt path), with a safe fallback so the
// file compiles on toolchains where the builtin is not declared.
// Builtin prototype (from hipcc diagnostic): arg0 = v4i in addrspace(1),
// arg1 = LDS-side pointer, then i32 offset, i32 cpol.
// ---------------------------------------------------------------------------
#if defined(__has_builtin)
#if __has_builtin(__builtin_amdgcn_global_load_async_to_lds_b128)
#define HAVE_ASYNC_LDS 1
#endif
#endif
#ifndef HAVE_ASYNC_LDS
#define HAVE_ASYNC_LDS 0
#endif

typedef __attribute__((address_space(1))) v4i* gptr_b128;
typedef __attribute__((address_space(3))) v4i* lptr_b128;

__device__ __forceinline__ void async_cp16(const _Float16* g, _Float16* l) {
#if HAVE_ASYNC_LDS
  __builtin_amdgcn_global_load_async_to_lds_b128(
      (gptr_b128)(void*)const_cast<_Float16*>(g),
      (lptr_b128)(void*)l, 0, 0);
#else
  *(v8h*)l = *(const v8h*)g;
#endif
}

__device__ __forceinline__ void wait_async_lds() {
#if HAVE_ASYNC_LDS
#if __has_builtin(__builtin_amdgcn_s_wait_asynccnt)
  __builtin_amdgcn_s_wait_asynccnt(0);
#else
  asm volatile("s_wait_asynccnt 0x0" ::: "memory");
#endif
#endif
}

// ---------------------------------------------------------------------------
// WMMA fragment loaders (layouts per CDNA5 ISA 7.12.2, wave32)
// A 16x32 f16: lane m=l&15; K chunks at [8h, 8h+8) and [16+8h, 16+8h+8)
// B 32x16 f16: lane n=l&15; K chunks at [8h, 8h+8) and [16+8h, 16+8h+8)
// ---------------------------------------------------------------------------
__device__ __forceinline__ v16h frag_a(const _Float16* base, int ld, int row0, int k0) {
  const int lane = threadIdx.x & 31;
  const int hlf  = lane >> 4;
  const _Float16* p = base + (size_t)(row0 + (lane & 15)) * ld + k0;
  v8h lo = *(const v8h*)(p + 8 * hlf);
  v8h hi = *(const v8h*)(p + 16 + 8 * hlf);
  v16h r;
#pragma unroll
  for (int i = 0; i < 8; ++i) { r[i] = lo[i]; r[i + 8] = hi[i]; }
  return r;
}

__device__ __forceinline__ v16h frag_b(const _Float16* base, int ld, int n0, int k0) {
  const int lane = threadIdx.x & 31;
  const int hlf  = lane >> 4;
  const _Float16* p = base + (size_t)(n0 + (lane & 15)) * ld + k0;
  v8h lo = *(const v8h*)(p + 8 * hlf);
  v8h hi = *(const v8h*)(p + 16 + 8 * hlf);
  v16h r;
#pragma unroll
  for (int i = 0; i < 8; ++i) { r[i] = lo[i]; r[i + 8] = hi[i]; }
  return r;
}

__device__ __forceinline__ v8f wmma_f16(v16h a, v16h b, v8f c) {
  return __builtin_amdgcn_wmma_f32_16x16x32_f16(false, a, false, b, (short)0, c,
                                                false, false);
}

// ---------------------------------------------------------------------------
// One-shot fp32 -> f16 cast
// ---------------------------------------------------------------------------
__global__ __launch_bounds__(256) void cvt_f32_f16(const float* __restrict__ in,
                                                   _Float16* __restrict__ out) {
  size_t base = ((size_t)blockIdx.x * 256 + threadIdx.x) * 8;
  float4 f0 = *(const float4*)(in + base);
  float4 f1 = *(const float4*)(in + base + 4);
  v8h h;
  h[0] = (_Float16)f0.x; h[1] = (_Float16)f0.y;
  h[2] = (_Float16)f0.z; h[3] = (_Float16)f0.w;
  h[4] = (_Float16)f1.x; h[5] = (_Float16)f1.y;
  h[6] = (_Float16)f1.z; h[7] = (_Float16)f1.w;
  *(v8h*)(out + base) = h;
}

// ---------------------------------------------------------------------------
// One-shot weight transpose+cast: W[k][n] fp32 -> Wt[n][k] f16 (LDS-tiled)
// ---------------------------------------------------------------------------
__global__ __launch_bounds__(256) void transpose_w_f16(const float* __restrict__ W,
                                                       _Float16* __restrict__ Wt) {
  __shared__ _Float16 T[64][72];
  const int t  = threadIdx.x;
  const int k0 = blockIdx.x * 64;
  const int n0 = blockIdx.y * 64;
#pragma unroll
  for (int i = 0; i < 4; ++i) {
    int idx = t + i * 256;                 // 1024 float4 chunks
    int kr = idx >> 4, c4 = idx & 15;
    float4 f = *(const float4*)&W[(size_t)(k0 + kr) * DIM + n0 + c4 * 4];
    T[c4 * 4 + 0][kr] = (_Float16)f.x;
    T[c4 * 4 + 1][kr] = (_Float16)f.y;
    T[c4 * 4 + 2][kr] = (_Float16)f.z;
    T[c4 * 4 + 3][kr] = (_Float16)f.w;
  }
  __syncthreads();
#pragma unroll
  for (int i = 0; i < 2; ++i) {
    int idx = t + i * 256;                 // 512 v8h chunks
    int nr = idx >> 3, c8 = idx & 7;
    *(v8h*)&Wt[(size_t)(n0 + nr) * DIM + k0 + c8 * 8] = *(const v8h*)&T[nr][c8 * 8];
  }
}

// ---------------------------------------------------------------------------
// 256x128-tile GEMM: C[8192 x 1024] = A_f16[8192 x 1024] * Bt_f16[n][k]
// 8 waves (4Mx2N), 64x64 per wave -> 16 WMMAs per 32-wide K step.
// Double-buffered LDS; next stage filled via async global->LDS while the
// current stage feeds the matrix pipes.
// EPI=0: f16 head-major [B*H, SEQ, 64] with scale.  EPI=1: fp32 + bias.
// ---------------------------------------------------------------------------
template <int EPI>
__global__ __launch_bounds__(256) void gemm256x128(const _Float16* __restrict__ Ah,
                                                   const _Float16* __restrict__ Bt,
                                                   _Float16* __restrict__ dstH,
                                                   float* __restrict__ dstF,
                                                   const float* __restrict__ bias,
                                                   float scale) {
  __shared__ _Float16 As[2][256][40];  // [m][k], 80B row stride (16B aligned)
  __shared__ _Float16 Bs[2][128][40];  // [n][k]

  const int t    = threadIdx.x;
  const int n0   = blockIdx.x * 128;
  const int m0   = blockIdx.y * 256;
  const int w    = t >> 5;
  const int lane = t & 31;
  const int wm   = (w >> 1) * 64;
  const int wn   = (w & 1) * 64;

  v8f acc[4][4] = {};

  auto issue_stage = [&](int kt, int buf) {
    const int k0 = kt * 32;
#pragma unroll
    for (int i = 0; i < 4; ++i) {        // A: 256 rows x 4 chunks = 1024
      int idx = t + i * 256;
      int row = idx >> 2, c8 = idx & 3;
      async_cp16(&Ah[(size_t)(m0 + row) * DIM + k0 + c8 * 8], &As[buf][row][c8 * 8]);
    }
#pragma unroll
    for (int i = 0; i < 2; ++i) {        // B: 128 rows x 4 chunks = 512
      int idx = t + i * 256;
      int row = idx >> 2, c8 = idx & 3;
      async_cp16(&Bt[(size_t)(n0 + row) * DIM + k0 + c8 * 8], &Bs[buf][row][c8 * 8]);
    }
  };

  issue_stage(0, 0);
  const int NKT = DIM / 32;
  for (int kt = 0; kt < NKT; ++kt) {
    const int cur = kt & 1;
    wait_async_lds();                    // stage `cur` landed in LDS
    __syncthreads();                     // visible to all; prev reads retired
    if (kt + 1 < NKT) issue_stage(kt + 1, cur ^ 1);  // overlap with compute

    v16h a[4];
#pragma unroll
    for (int i = 0; i < 4; ++i) a[i] = frag_a(&As[cur][0][0], 40, wm + i * 16, 0);
#pragma unroll
    for (int j = 0; j < 4; ++j) {
      v16h b = frag_b(&Bs[cur][0][0], 40, wn + j * 16, 0);
#pragma unroll
      for (int i = 0; i < 4; ++i) acc[i][j] = wmma_f16(a[i], b, acc[i][j]);
    }
  }

  // ---- epilogue ----
  const int hlf = lane >> 4, l16 = lane & 15;
#pragma unroll
  for (int i = 0; i < 4; ++i)
#pragma unroll
    for (int j = 0; j < 4; ++j)
#pragma unroll
      for (int e = 0; e < 8; ++e) {
        int r = m0 + wm + i * 16 + e + 8 * hlf;
        int c = n0 + wn + j * 16 + l16;
        float val = acc[i][j][e];
        if (EPI == 0) {
          int b = r >> 11, tok = r & (SEQ - 1);
          int h = c >> 6, d = c & (HEAD_DIM - 1);
          dstH[(size_t)(b * HEADS + h) * (SEQ * HEAD_DIM) + tok * HEAD_DIM + d] =
              (_Float16)(val * scale);
        } else {
          dstF[(size_t)r * DIM + c] = val + bias[c];
        }
      }
}

// ---------------------------------------------------------------------------
// Flash attention: block = 128 thr (4 waves), one (b,h) x 64 query rows.
// ---------------------------------------------------------------------------
__global__ __launch_bounds__(128) void attn_kernel(const _Float16* __restrict__ q,
                                                   const _Float16* __restrict__ k,
                                                   const _Float16* __restrict__ v,
                                                   _Float16* __restrict__ o) {
  __shared__ _Float16 Qs[64][72];        // [qrow][d]
  __shared__ _Float16 Ks[64][72];        // [key][d]
  __shared__ _Float16 Vt[64][72];        // [d][key]
  __shared__ _Float16 Ps[4][16][72];     // per-wave P tile [qrow][key]

  const int t    = threadIdx.x;
  const int wave = t >> 5;
  const int lane = t & 31;
  const int hlf  = lane >> 4, l16 = lane & 15;
  const int bh   = blockIdx.y;
  const int qc   = blockIdx.x;

  const size_t headBase = (size_t)bh * SEQ * HEAD_DIM;
  const _Float16* qp = q + headBase + (size_t)qc * 64 * HEAD_DIM;

#pragma unroll
  for (int i = 0; i < 4; ++i) {          // Q: 512 chunks of 8 halfs (async)
    int idx = t + i * 128;
    int row = idx >> 3, c8 = idx & 7;
    async_cp16(&qp[row * HEAD_DIM + c8 * 8], &Qs[row][c8 * 8]);
  }

  v8f accO[4] = {};
  float Mrow[8], Lrow[8];
#pragma unroll
  for (int i = 0; i < 8; ++i) { Mrow[i] = -1e30f; Lrow[i] = 0.f; }

  for (int kb = 0; kb < SEQ / 64; ++kb) {
    __syncthreads();                     // prior iter done reading Ks/Vt
    const _Float16* kp = k + headBase + (size_t)kb * 64 * HEAD_DIM;
    const _Float16* vp = v + headBase + (size_t)kb * 64 * HEAD_DIM;
#pragma unroll
    for (int i = 0; i < 4; ++i) {
      int idx = t + i * 128;
      int row = idx >> 3, c8 = idx & 7;
      async_cp16(&kp[row * HEAD_DIM + c8 * 8], &Ks[row][c8 * 8]);  // straight copy
      v8h vld = *(const v8h*)&vp[row * HEAD_DIM + c8 * 8];         // V transpose
#pragma unroll
      for (int j = 0; j < 8; ++j) Vt[c8 * 8 + j][row] = vld[j];
    }
    wait_async_lds();
    __syncthreads();

    // ---- S(16x64) = Q * K^T (contraction over d=64) ----
    v8f s[4] = {};
#pragma unroll
    for (int ks = 0; ks < HEAD_DIM; ks += 32) {
      v16h aq = frag_a(&Qs[0][0], 72, wave * 16, ks);
#pragma unroll
      for (int j = 0; j < 4; ++j) {
        v16h bk = frag_b(&Ks[0][0], 72, j * 16, ks);
        s[j] = wmma_f16(aq, bk, s[j]);
      }
    }

    // ---- online softmax over this 64-key block ----
#pragma unroll
    for (int e = 0; e < 8; ++e) {        // row m = e + 8*hlf
      float cur = fmaxf(fmaxf(s[0][e], s[1][e]), fmaxf(s[2][e], s[3][e]));
      cur = fmaxf(cur, __shfl_xor(cur, 1, 32));
      cur = fmaxf(cur, __shfl_xor(cur, 2, 32));
      cur = fmaxf(cur, __shfl_xor(cur, 4, 32));
      cur = fmaxf(cur, __shfl_xor(cur, 8, 32));
      float mnew = fmaxf(Mrow[e], cur);
      float p0 = __expf(s[0][e] - mnew), p1 = __expf(s[1][e] - mnew);
      float p2 = __expf(s[2][e] - mnew), p3 = __expf(s[3][e] - mnew);
      float sum = p0 + p1 + p2 + p3;
      sum += __shfl_xor(sum, 1, 32);
      sum += __shfl_xor(sum, 2, 32);
      sum += __shfl_xor(sum, 4, 32);
      sum += __shfl_xor(sum, 8, 32);
      float fac = __expf(Mrow[e] - mnew);
      Mrow[e] = mnew;
      Lrow[e] = Lrow[e] * fac + sum;
#pragma unroll
      for (int j = 0; j < 4; ++j) accO[j][e] *= fac;
      int prow = e + 8 * hlf;
      Ps[wave][prow][0 * 16 + l16] = (_Float16)p0;
      Ps[wave][prow][1 * 16 + l16] = (_Float16)p1;
      Ps[wave][prow][2 * 16 + l16] = (_Float16)p2;
      Ps[wave][prow][3 * 16 + l16] = (_Float16)p3;
    }

    // ---- O(16x64) += P * V (contraction over 64 keys) ----
#pragma unroll
    for (int ks = 0; ks < 64; ks += 32) {
      v16h ap = frag_a(&Ps[wave][0][0], 72, 0, ks);
#pragma unroll
      for (int j = 0; j < 4; ++j) {
        v16h bv = frag_b(&Vt[0][0], 72, j * 16, ks);
        accO[j] = wmma_f16(ap, bv, accO[j]);
      }
    }
  }

  // ---- normalize and store O as f16 [B*N, HEADS*64] ----
  const int b = bh >> 4, h = bh & (HEADS - 1);
#pragma unroll
  for (int e = 0; e < 8; ++e) {
    float rcp = 1.0f / Lrow[e];
    int tok = qc * 64 + wave * 16 + e + 8 * hlf;
#pragma unroll
    for (int j = 0; j < 4; ++j) {
      int d = j * 16 + l16;
      o[(size_t)(b * SEQ + tok) * DIM + h * HEAD_DIM + d] =
          (_Float16)(accO[j][e] * rcp);
    }
  }
}

// ---------------------------------------------------------------------------
extern "C" void kernel_launch(void* const* d_in, const int* in_sizes, int n_in,
                              void* d_out, int out_size, void* d_ws, size_t ws_size,
                              hipStream_t stream) {
  const float* x  = (const float*)d_in[0];
  const float* Wq = (const float*)d_in[1];
  const float* Wk = (const float*)d_in[2];
  const float* Wv = (const float*)d_in[3];
  const float* Wo = (const float*)d_in[4];
  const float* bo = (const float*)d_in[5];
  float* out = (float*)d_out;

  // workspace layout (f16): xh | Q | K | V | O (16 MB each) | 4 transposed W
  const size_t MAT = (size_t)MROWS * DIM;   // 8M elements
  const size_t WSZ = (size_t)DIM * DIM;     // 1M elements
  _Float16* xh  = (_Float16*)d_ws;
  _Float16* qh  = xh + MAT;
  _Float16* kh  = qh + MAT;
  _Float16* vh  = kh + MAT;
  _Float16* oh  = vh + MAT;
  _Float16* wqt = oh + MAT;
  _Float16* wkt = wqt + WSZ;
  _Float16* wvt = wkt + WSZ;
  _Float16* wot = wvt + WSZ;

  // one-shot precision/layout prep (~50 MB extra traffic, ~2us at 23.3 TB/s)
  cvt_f32_f16<<<(int)(MAT / (256 * 8)), 256, 0, stream>>>(x, xh);
  dim3 tg(DIM / 64, DIM / 64);
  transpose_w_f16<<<tg, 256, 0, stream>>>(Wq, wqt);
  transpose_w_f16<<<tg, 256, 0, stream>>>(Wk, wkt);
  transpose_w_f16<<<tg, 256, 0, stream>>>(Wv, wvt);
  transpose_w_f16<<<tg, 256, 0, stream>>>(Wo, wot);

  dim3 g(DIM / 128, MROWS / 256);
  const float qscale = 0.125f;  // 1/sqrt(HEAD_DIM), folded into Q
  gemm256x128<0><<<g, 256, 0, stream>>>(xh, wqt, qh, nullptr, nullptr, qscale);
  gemm256x128<0><<<g, 256, 0, stream>>>(xh, wkt, kh, nullptr, nullptr, 1.0f);
  gemm256x128<0><<<g, 256, 0, stream>>>(xh, wvt, vh, nullptr, nullptr, 1.0f);

  attn_kernel<<<dim3(SEQ / 64, BATCH * HEADS), 128, 0, stream>>>(qh, kh, vh, oh);

  gemm256x128<1><<<g, 256, 0, stream>>>(oh, wot, nullptr, out, bo, 1.0f);

  (void)in_sizes; (void)n_in; (void)out_size; (void)ws_size;
}
